// GAB_24146306138543
// MI455X (gfx1250) — matrix operand order repaired and checked by reading, not compiled, not run on hardware
//
#include <hip/hip_runtime.h>

// ---------------------------------------------------------------------------
// CDNA5 (gfx1250) implementation of a CrossFormer grid-attention block.
// All heavy math goes through v_wmma_f32_16x16x32_f16 (f16 in, f32 accum).
// ---------------------------------------------------------------------------

typedef __attribute__((ext_vector_type(16))) _Float16 v16h;
typedef __attribute__((ext_vector_type(8)))  float    v8f;

union V16H { v16h v; _Float16 e[16]; };
union V8F  { v8f  v; float    e[8];  };

__device__ __forceinline__ v8f wmma16(const v16h& a, const v16h& b, const v8f& c) {
  // D = A(16x32 f16) * B(32x16 f16) + C(16x16 f32)
  return __builtin_amdgcn_wmma_f32_16x16x32_f16(false, a, false, b, (short)0, c,
                                                false, false);
}

__device__ __forceinline__ v8f v8f_zero() {
  v8f z = {0.f, 0.f, 0.f, 0.f, 0.f, 0.f, 0.f, 0.f};
  return z;
}

// A fragment: 16x32 f16, rows striped per lane; element s of lane half h holds k = 16h + s.
// p points at (row0, k0); stride in halves.
__device__ __forceinline__ v16h load_a_frag(const _Float16* p, int stride, int lane) {
  int l = lane & 15;
  int hb = (lane >> 4) * 16;
  const _Float16* r = p + l * stride;
  V16H a;
#pragma unroll
  for (int i = 0; i < 4; ++i) {
    int k = hb + 2 * i;
    a.e[2 * i]     = r[k];
    a.e[2 * i + 1] = r[k + 1];
    a.e[8 + 2 * i] = r[8 + k];
    a.e[9 + 2 * i] = r[8 + k + 1];
  }
  return a.v;
}

// B fragment from row-major KxN storage: B[k][n] = p[k*stride + n].
__device__ __forceinline__ v16h load_b_frag(const _Float16* p, int stride, int lane) {
  int n  = lane & 15;
  int hb = (lane >> 4) * 16;
  const _Float16* r = p + hb * stride + n;
  V16H b;
#pragma unroll
  for (int i = 0; i < 16; ++i) b.e[i] = r[i * stride];
  return b.v;
}

// B fragment from "transposed" source (tokens x d): B[k][n] = p[n*stride + k].
__device__ __forceinline__ v16h load_bT_frag(const _Float16* p, int stride, int lane) {
  int n  = lane & 15;
  int hb = (lane >> 4) * 16;
  const _Float16* r = p + n * stride + hb;
  V16H b;
#pragma unroll
  for (int i = 0; i < 16; ++i) b.e[i] = r[i];
  return b.v;
}

// ---------------------------------------------------------------------------
// LayerNorm over 192 channels: fp32 in -> f16 out. One wave per row.
// ---------------------------------------------------------------------------
__global__ __launch_bounds__(256) void ln_kernel(const float* __restrict__ x,
                                                 const float* __restrict__ g,
                                                 const float* __restrict__ b,
                                                 _Float16* __restrict__ out) {
  int lane = threadIdx.x & 31;
  int row  = blockIdx.x * 8 + (threadIdx.x >> 5);
  const float* xr = x + (size_t)row * 192;
  float v[6];
  float s = 0.f;
#pragma unroll
  for (int i = 0; i < 6; ++i) { v[i] = xr[lane + 32 * i]; s += v[i]; }
#pragma unroll
  for (int m = 1; m < 32; m <<= 1) s += __shfl_xor(s, m, 32);
  float mu = s * (1.f / 192.f);
  float var = 0.f;
#pragma unroll
  for (int i = 0; i < 6; ++i) { float d = v[i] - mu; var += d * d; }
#pragma unroll
  for (int m = 1; m < 32; m <<= 1) var += __shfl_xor(var, m, 32);
  var *= (1.f / 192.f);
  float rs = rsqrtf(var + 1e-5f);
  _Float16* orow = out + (size_t)row * 192;
#pragma unroll
  for (int i = 0; i < 6; ++i) {
    int c = lane + 32 * i;
    orow[c] = (_Float16)((v[i] - mu) * rs * g[c] + b[c]);
  }
}

// ---------------------------------------------------------------------------
// One-shot f32 -> f16 weight conversion (hoisted out of the GEMM K-loop).
// ---------------------------------------------------------------------------
__global__ __launch_bounds__(256) void cvt_kernel(const float* __restrict__ src,
                                                  _Float16* __restrict__ dst,
                                                  int n4) {
  int i = blockIdx.x * 256 + threadIdx.x;
  if (i >= n4) return;
  float4 v = ((const float4*)src)[i];
  union { _Float16 h[4]; float2 f; } u;
  u.h[0] = (_Float16)v.x; u.h[1] = (_Float16)v.y;
  u.h[2] = (_Float16)v.z; u.h[3] = (_Float16)v.w;
  ((float2*)dst)[i] = u.f;
}

// ---------------------------------------------------------------------------
// Generic WMMA GEMM: C[M,N] = A[M,K](f16) * W[N,K](f16)^T + bias
// MODE 0: store f16   MODE 1: exact-GELU then f16   MODE 2: +res, store f32
// Block: 256 threads (8 waves). Tile 128x64, K-step 32; each wave owns 32x32
// (2 A-frags x 2 B-frags -> 4 WMMAs per K-step).
// ---------------------------------------------------------------------------
#define BM 128
#define BN 64
#define BK 32

template <int MODE>
__global__ __launch_bounds__(256) void gemm_kernel(
    const _Float16* __restrict__ A, const _Float16* __restrict__ W,
    const float* __restrict__ bias, const float* __restrict__ res,
    void* __restrict__ outp, int M, int N, int K) {
  __shared__ _Float16 As[BM * BK];
  __shared__ _Float16 Bs[BK * BN];
  int tid = threadIdx.x, lane = tid & 31, wave = tid >> 5;
  int m0 = blockIdx.y * BM, n0 = blockIdx.x * BN;
  int r0 = (wave & 3) * 32;
  int c0 = (wave >> 2) * 32;

  V8F acc[2][2];
#pragma unroll
  for (int i = 0; i < 2; ++i)
#pragma unroll
    for (int j = 0; j < 2; ++j) acc[i][j].v = v8f_zero();

  int ra  = tid >> 1, cha = (tid & 1) * 16;   // A: 16 halves / thread
  int nb  = tid >> 2, chb = (tid & 3) * 8;    // B: 8 halves / thread

  for (int k0 = 0; k0 < K; k0 += BK) {
    {  // A tile: 128x32 halves
      const float4* s4 =
          reinterpret_cast<const float4*>(A + (size_t)(m0 + ra) * K + k0 + cha);
      float4* d4 = reinterpret_cast<float4*>(&As[ra * BK + cha]);
      d4[0] = s4[0];
      d4[1] = s4[1];
    }
    {  // W tile (f16), stored K-major into Bs[k][n]
      const float4 wv =
          *reinterpret_cast<const float4*>(W + (size_t)(n0 + nb) * K + k0 + chb);
      const _Float16* wh = reinterpret_cast<const _Float16*>(&wv);
#pragma unroll
      for (int i = 0; i < 8; ++i) Bs[(chb + i) * BN + nb] = wh[i];
    }
    if (k0 + BK < K) {  // pull next K-panel toward cache (global_prefetch_b8)
      __builtin_prefetch(A + (size_t)(m0 + ra) * K + k0 + BK + cha, 0, 1);
      __builtin_prefetch(W + (size_t)(n0 + nb) * K + k0 + BK + chb, 0, 1);
    }
    __syncthreads();
    v16h a0 = load_a_frag(&As[r0 * BK], BK, lane);
    v16h a1 = load_a_frag(&As[(r0 + 16) * BK], BK, lane);
    v16h b0 = load_b_frag(&Bs[c0], BN, lane);
    v16h b1 = load_b_frag(&Bs[c0 + 16], BN, lane);
    acc[0][0].v = wmma16(a0, b0, acc[0][0].v);
    acc[0][1].v = wmma16(a0, b1, acc[0][1].v);
    acc[1][0].v = wmma16(a1, b0, acc[1][0].v);
    acc[1][1].v = wmma16(a1, b1, acc[1][1].v);
    __syncthreads();
  }

  int nlane = lane & 15;
  int mo    = (lane >> 4) * 8;
#pragma unroll
  for (int rt = 0; rt < 2; ++rt) {
#pragma unroll
    for (int t = 0; t < 2; ++t) {
      V8F& a  = acc[rt][t];
      int col = n0 + c0 + t * 16 + nlane;
      float bv = bias[col];
#pragma unroll
      for (int j = 0; j < 8; ++j) {
        int m = m0 + r0 + rt * 16 + mo + j;
        float v = a.e[j] + bv;
        size_t off = (size_t)m * N + col;
        if (MODE == 0) {
          ((_Float16*)outp)[off] = (_Float16)v;
        } else if (MODE == 1) {
          v = 0.5f * v * (1.0f + erff(v * 0.70710678118654752f));
          ((_Float16*)outp)[off] = (_Float16)v;
        } else {
          ((float*)outp)[off] = v + res[off];
        }
      }
    }
  }
}

// ---------------------------------------------------------------------------
// DynamicPosBias tiny MLP. One thread per table row (961 rows, 2 stages).
// Leaf order = JAX dict pytree (alphabetical):
// 0 l1_b 1 l1_w 2 l2_b 3 l2_w 4 l3_b 5 l3_w 6 ln1_b 7 ln1_g 8 ln2_b 9 ln2_g
// 10 ln3_b 11 ln3_g 12 pp_b 13 pp_w
// ---------------------------------------------------------------------------
struct PosP { const float* p[14]; };

__global__ void posmlp_kernel(PosP P0, PosP P1, float* __restrict__ tab) {
  int r = blockIdx.x * blockDim.x + threadIdx.x;
  if (r >= 961) return;
  PosP P = blockIdx.y ? P1 : P0;
  float b0 = (float)(r / 31) - 15.0f;
  float b1 = (float)(r % 31) - 15.0f;
  float h[12], t[12];
  const float* ppw = P.p[13];
  const float* ppb = P.p[12];
#pragma unroll
  for (int j = 0; j < 12; ++j) h[j] = ppw[2 * j] * b0 + ppw[2 * j + 1] * b1 + ppb[j];
  const float* lng[3] = {P.p[7], P.p[9], P.p[11]};
  const float* lnb[3] = {P.p[6], P.p[8], P.p[10]};
  const float* lw[3]  = {P.p[1], P.p[3], P.p[5]};
  const float* lb[3]  = {P.p[0], P.p[2], P.p[4]};
  for (int s = 0; s < 3; ++s) {
    float mu = 0.f;
#pragma unroll
    for (int j = 0; j < 12; ++j) mu += h[j];
    mu *= (1.f / 12.f);
    float var = 0.f;
#pragma unroll
    for (int j = 0; j < 12; ++j) { float d = h[j] - mu; var += d * d; }
    var *= (1.f / 12.f);
    float rs = rsqrtf(var + 1e-5f);
#pragma unroll
    for (int j = 0; j < 12; ++j) {
      float u = (h[j] - mu) * rs * lng[s][j] + lnb[s][j];
      t[j] = u > 0.f ? u : 0.f;
    }
    int od = (s == 2) ? 6 : 12;
    for (int j = 0; j < od; ++j) {
      float o = lb[s][j];
#pragma unroll
      for (int k = 0; k < 12; ++k) o += lw[s][j * 12 + k] * t[k];
      h[j] = o;
    }
  }
  float* dst = tab + ((size_t)blockIdx.y * 961 + r) * 6;
#pragma unroll
  for (int j = 0; j < 6; ++j) dst[j] = h[j];
}

// Expand (961,6) table -> (stage, head, 256, 256) bias using rel-pos index.
__global__ void biasexpand_kernel(const float* __restrict__ tab,
                                  float* __restrict__ bias) {
  int idx = blockIdx.x * blockDim.x + threadIdx.x;
  if (idx >= 2 * 6 * 65536) return;
  int j  = idx & 255;
  int i  = (idx >> 8) & 255;
  int hh = (idx >> 16) % 6;
  int st = (idx >> 16) / 6;
  int dr = (i >> 4) - (j >> 4) + 15;
  int dc = (i & 15) - (j & 15) + 15;
  int t  = dr * 31 + dc;
  bias[idx] = tab[((size_t)st * 961 + t) * 6 + hh];
}

// ---------------------------------------------------------------------------
// Two-stage grid attention: one block per (window, head). 256 threads.
// Each wave owns 32 query rows. hd=32 => one WMMA per 16x16 logits tile.
// ---------------------------------------------------------------------------
__device__ __forceinline__ void attn_stage(
    const _Float16* __restrict__ Qs,   // 256x32 LDS (queries)
    const _Float16* __restrict__ Ks,   // 256x32 LDS (keys, row-major tokens)
    const _Float16* __restrict__ Vs,   // 256x32 LDS (values)
    const float* __restrict__ biasH,   // (256,256) f32, this head
    _Float16* __restrict__ Ps,         // 256x256 LDS scratch
    int wave, int lane, V8F oacc[2][2]) {
  const float scale = 0.17677669529663687f;  // 32^-0.5
  int nlane = lane & 15;
  int mo    = (lane >> 4) * 8;

#pragma unroll
  for (int rt = 0; rt < 2; ++rt) {
    int r0 = wave * 32 + rt * 16;
    v16h af = load_a_frag(Qs + r0 * 32, 32, lane);
    V8F s[16];
#pragma unroll
    for (int ct = 0; ct < 16; ++ct) {
      v16h bf = load_bT_frag(Ks + ct * 16 * 32, 32, lane);
      s[ct].v = wmma16(af, bf, v8f_zero());
    }
    // scale + bias, row max
    float mx[8];
#pragma unroll
    for (int j = 0; j < 8; ++j) mx[j] = -1e30f;
#pragma unroll
    for (int ct = 0; ct < 16; ++ct) {
#pragma unroll
      for (int j = 0; j < 8; ++j) {
        int m = r0 + mo + j;
        int col = ct * 16 + nlane;
        float v = s[ct].e[j] * scale + biasH[(size_t)m * 256 + col];
        s[ct].e[j] = v;
        mx[j] = fmaxf(mx[j], v);
      }
    }
#pragma unroll
    for (int j = 0; j < 8; ++j) {
#pragma unroll
      for (int d = 1; d < 16; d <<= 1) mx[j] = fmaxf(mx[j], __shfl_xor(mx[j], d, 32));
    }
    float sm[8];
#pragma unroll
    for (int j = 0; j < 8; ++j) sm[j] = 0.f;
#pragma unroll
    for (int ct = 0; ct < 16; ++ct) {
#pragma unroll
      for (int j = 0; j < 8; ++j) {
        float e = __expf(s[ct].e[j] - mx[j]);
        s[ct].e[j] = e;
        sm[j] += e;
      }
    }
#pragma unroll
    for (int j = 0; j < 8; ++j) {
#pragma unroll
      for (int d = 1; d < 16; d <<= 1) sm[j] += __shfl_xor(sm[j], d, 32);
      sm[j] = 1.f / sm[j];
    }
#pragma unroll
    for (int ct = 0; ct < 16; ++ct) {
#pragma unroll
      for (int j = 0; j < 8; ++j) {
        int m = r0 + mo + j;
        Ps[(size_t)m * 256 + ct * 16 + nlane] = (_Float16)(s[ct].e[j] * sm[j]);
      }
    }
  }
  // P @ V — each wave reads only its own Ps rows: no block barrier needed.
#pragma unroll
  for (int rt = 0; rt < 2; ++rt) {
    int r0 = wave * 32 + rt * 16;
#pragma unroll
    for (int dt = 0; dt < 2; ++dt) {
      v8f a = v8f_zero();
#pragma unroll
      for (int ks = 0; ks < 8; ++ks) {
        v16h af = load_a_frag(Ps + (size_t)r0 * 256 + ks * 32, 256, lane);
        v16h bf = load_b_frag(Vs + ks * 32 * 32 + dt * 16, 32, lane);
        a = wmma16(af, bf, a);
      }
      oacc[rt][dt].v = a;
    }
  }
}

__global__ __launch_bounds__(256) void attn_kernel(
    const _Float16* __restrict__ qkv,  // (32768, 576) f16
    const _Float16* __restrict__ xn,   // (32768, 192) f16 (LN1 output)
    const float* __restrict__ bias1,   // (6,256,256)
    const float* __restrict__ bias2,   // (6,256,256)
    _Float16* __restrict__ out) {      // (32768, 192) f16, window-reversed
  extern __shared__ _Float16 smem[];
  _Float16* g_s  = smem;              // 256x32 grid (stage-1 Q / stage-2 K)
  _Float16* k_s  = g_s + 8192;
  _Float16* v_s  = k_s + 8192;
  _Float16* q_s  = v_s + 8192;
  _Float16* x1_s = q_s + 8192;        // stage-1 output (stage-2 V)
  _Float16* p_s  = x1_s + 8192;       // 256x256 probabilities

  int head = blockIdx.x, win = blockIdx.y;
  int tid = threadIdx.x, lane = tid & 31, wave = tid >> 5;
  int b = win >> 6, wr = (win >> 3) & 7, wc = win & 7;

  {  // cooperative tile loads: thread == token
    int tok = tid;
    int gRow = b * 16384 + ((wr << 4) + (tok >> 4)) * 128 + (wc << 4) + (tok & 15);
    int gridRow = b * 16384 + ((tok >> 4) * 8 + wr) * 128 + ((tok & 15) * 8 + wc);
    const float4* qsrc = (const float4*)(qkv + (size_t)gRow * 576 + head * 32);
    const float4* ksrc = (const float4*)(qkv + (size_t)gRow * 576 + 192 + head * 32);
    const float4* vsrc = (const float4*)(qkv + (size_t)gRow * 576 + 384 + head * 32);
    const float4* gsrc = (const float4*)(xn + (size_t)gridRow * 192 + head * 32);
    float4* qd = (float4*)(q_s + tok * 32);
    float4* kd = (float4*)(k_s + tok * 32);
    float4* vd = (float4*)(v_s + tok * 32);
    float4* gd = (float4*)(g_s + tok * 32);
#pragma unroll
    for (int c = 0; c < 4; ++c) {
      qd[c] = qsrc[c]; kd[c] = ksrc[c]; vd[c] = vsrc[c]; gd[c] = gsrc[c];
    }
  }
  __syncthreads();

  int nlane = lane & 15;
  int mo    = (lane >> 4) * 8;

  // Stage 1: x1 = softmax(scale * grid @ k^T + b1) @ v
  V8F o1[2][2];
  attn_stage(g_s, k_s, v_s, bias1 + (size_t)head * 65536, p_s, wave, lane, o1);
#pragma unroll
  for (int rt = 0; rt < 2; ++rt)
#pragma unroll
    for (int dt = 0; dt < 2; ++dt)
#pragma unroll
      for (int j = 0; j < 8; ++j) {
        int m = wave * 32 + rt * 16 + mo + j;
        x1_s[m * 32 + dt * 16 + nlane] = (_Float16)o1[rt][dt].e[j];
      }
  __syncthreads();

  // Stage 2: x2 = softmax(scale * q @ grid^T + b2) @ x1
  V8F o2[2][2];
  attn_stage(q_s, g_s, x1_s, bias2 + (size_t)head * 65536, p_s, wave, lane, o2);

  // window-reverse scatter to (B*N, 192) f16
#pragma unroll
  for (int rt = 0; rt < 2; ++rt)
#pragma unroll
    for (int dt = 0; dt < 2; ++dt)
#pragma unroll
      for (int j = 0; j < 8; ++j) {
        int m = wave * 32 + rt * 16 + mo + j;
        int gRow = b * 16384 + ((wr << 4) + (m >> 4)) * 128 + (wc << 4) + (m & 15);
        out[(size_t)gRow * 192 + head * 32 + dt * 16 + nlane] =
            (_Float16)o2[rt][dt].e[j];
      }
}

// ---------------------------------------------------------------------------
extern "C" void kernel_launch(void* const* d_in, const int* in_sizes, int n_in,
                              void* d_out, int out_size, void* d_ws, size_t ws_size,
                              hipStream_t stream) {
  (void)in_sizes; (void)n_in; (void)out_size; (void)ws_size;

  const float* x      = (const float*)d_in[0];
  const float* qkv_w  = (const float*)d_in[1];
  const float* qkv_b  = (const float*)d_in[2];
  const float* proj_w = (const float*)d_in[3];
  const float* proj_b = (const float*)d_in[4];
  const float* n1g    = (const float*)d_in[5];
  const float* n1b    = (const float*)d_in[6];
  const float* n2g    = (const float*)d_in[7];
  const float* n2b    = (const float*)d_in[8];
  const float* fc1_w  = (const float*)d_in[9];
  const float* fc1_b  = (const float*)d_in[10];
  const float* fc2_w  = (const float*)d_in[11];
  const float* fc2_b  = (const float*)d_in[12];
  PosP p1, p2;
  for (int i = 0; i < 14; ++i) {
    p1.p[i] = (const float*)d_in[13 + i];
    p2.p[i] = (const float*)d_in[27 + i];
  }

  const int M = 32768;  // B * H * W tokens
  char* w = (char*)d_ws;
  _Float16* xn    = (_Float16*)w; w += (size_t)M * 192 * 2;   // 12.6 MB
  _Float16* qkvo  = (_Float16*)w; w += (size_t)M * 576 * 2;   // 37.7 MB
  _Float16* attn  = (_Float16*)w; w += (size_t)M * 192 * 2;   // 12.6 MB
  float*    xmid  = (float*)w;    w += (size_t)M * 192 * 4;   // 25.2 MB
  _Float16* hbuf  = (_Float16*)w; w += (size_t)M * 384 * 2;   // 25.2 MB
  float*    tab   = (float*)w;    w += 49152;                  // 2*(961,6) padded
  float*    biasT = (float*)w;    w += (size_t)2 * 6 * 65536 * 4;
  _Float16* wqh   = (_Float16*)w; w += (size_t)576 * 192 * 2;  // f16 weights
  _Float16* wph   = (_Float16*)w; w += (size_t)192 * 192 * 2;
  _Float16* w1h   = (_Float16*)w; w += (size_t)384 * 192 * 2;
  _Float16* w2h   = (_Float16*)w; w += (size_t)192 * 384 * 2;

  // 0) weights -> f16 (once per launch; tiny)
  cvt_kernel<<<dim3(108), 256, 0, stream>>>(qkv_w, wqh, 576 * 192 / 4);
  cvt_kernel<<<dim3(36), 256, 0, stream>>>(proj_w, wph, 192 * 192 / 4);
  cvt_kernel<<<dim3(72), 256, 0, stream>>>(fc1_w, w1h, 384 * 192 / 4);
  cvt_kernel<<<dim3(72), 256, 0, stream>>>(fc2_w, w2h, 192 * 384 / 4);

  // 1) LN1 -> xn (f16)
  ln_kernel<<<dim3(M / 8), 256, 0, stream>>>(x, n1g, n1b, xn);
  // 2) QKV GEMM: (M,576) f16
  gemm_kernel<0><<<dim3(576 / BN, M / BM), 256, 0, stream>>>(
      xn, wqh, qkv_b, nullptr, qkvo, M, 576, 192);
  // 3) bias tables
  posmlp_kernel<<<dim3(4, 2), 256, 0, stream>>>(p1, p2, tab);
  biasexpand_kernel<<<dim3((2 * 6 * 65536) / 256), 256, 0, stream>>>(tab, biasT);
  // 4) two-stage attention; 208 KB dynamic LDS per block (<= 320 KB/WGP)
  attn_kernel<<<dim3(6, 128), 256, 212992, stream>>>(
      qkvo, xn, biasT, biasT + 6 * 65536, attn);
  // 5) proj + residual -> xmid (f32)
  gemm_kernel<2><<<dim3(192 / BN, M / BM), 256, 0, stream>>>(
      attn, wph, proj_b, x, xmid, M, 192, 192);
  // 6) LN2 -> reuse xn buffer (f16)
  ln_kernel<<<dim3(M / 8), 256, 0, stream>>>(xmid, n2g, n2b, xn);
  // 7) fc1 + exact GELU -> hbuf (f16)
  gemm_kernel<1><<<dim3(384 / BN, M / BM), 256, 0, stream>>>(
      xn, w1h, fc1_b, nullptr, hbuf, M, 384, 192);
  // 8) fc2 + residual -> d_out (f32)
  gemm_kernel<2><<<dim3(192 / BN, M / BM), 256, 0, stream>>>(
      hbuf, w2h, fc2_b, xmid, d_out, M, 192, 384);
}
